// MultiheadAttention_20495583937213
// MI455X (gfx1250) — matrix operand order repaired and checked by reading, not compiled, not run on hardware
//
#include <hip/hip_runtime.h>

typedef _Float16 f16;
typedef __attribute__((ext_vector_type(16))) _Float16     v16h;
typedef __attribute__((ext_vector_type(8)))  _Float16     v8h;
typedef __attribute__((ext_vector_type(8)))  float        v8f;
typedef __attribute__((ext_vector_type(4)))  float        v4f;
typedef __attribute__((ext_vector_type(4)))  unsigned int u32x4;
typedef __attribute__((ext_vector_type(8)))  int          i32x8;
typedef __attribute__((ext_vector_type(4)))  int          i32x4;

constexpr int Tn = 2048, Bn = 2, Dn = 1024, Hn = 16, HDn = 64, En = 1024;
constexpr int Mn = Tn * Bn;                       // 4096 rows for projections

#if defined(__gfx1250__) && __has_builtin(__builtin_amdgcn_tensor_load_to_lds)
#define MHA_USE_TDM 1
#else
#define MHA_USE_TDM 0
#endif

static __device__ __forceinline__ v16h cat8(v8h lo, v8h hi) {
    return __builtin_shufflevector(lo, hi, 0,1,2,3,4,5,6,7,8,9,10,11,12,13,14,15);
}
static __device__ __forceinline__ v8f wmma_f16(v16h a, v16h b, v8f c) {
    return __builtin_amdgcn_wmma_f32_16x16x32_f16(false, a, false, b, (short)0, c, false, false);
}
static __device__ __forceinline__ v8f zero8() {
    v8f z = {0.f,0.f,0.f,0.f,0.f,0.f,0.f,0.f};
    return z;
}

// ---------------------------------------------------------------------------
// Kernel 1: QKV projection (templated on P so the epilogue is branch-free).
//   P=0: q -> qh[b][h][t][hd] * HD^-0.5   (f16)
//   P=1: k -> kh[b][h][t][hd]             (f16)
//   P=2: v -> vT[b][h][hd][t]             (f16, transposed for PV B-fragments)
// Tile: BM=64, BN=64, BK=32; 128 threads = 4 waves, 16 M-rows per wave.
// ---------------------------------------------------------------------------
template <int P>
__global__ __launch_bounds__(128)
void mha_qkv_proj(const float* __restrict__ X, const float* __restrict__ Win,
                  const float* __restrict__ bin, f16* __restrict__ dst)
{
    const float* __restrict__ Wp = Win + (size_t)P * En * Dn;

    const int mBase = blockIdx.x * 64;
    const int nBase = blockIdx.y * 64;

    __shared__ __align__(16) f16 lX[64][40];
    __shared__ __align__(16) f16 lW[64][40];

    const int tid  = threadIdx.x;
    const int wv   = tid >> 5;
    const int lane = tid & 31;
    const int lm   = lane & 15;
    const int lg   = lane >> 4;

    v8f acc[4];
    #pragma unroll
    for (int f = 0; f < 4; ++f) acc[f] = zero8();

    for (int kk = 0; kk < Dn; kk += 32) {
        __syncthreads();
        #pragma unroll
        for (int j = 0; j < 4; ++j) {                 // 512 float4 chunks / 128 thr
            int c   = tid + j * 128;
            int row = c >> 3;
            int col = (c & 7) << 2;
            v4f xv  = *(const v4f*)(X  + (size_t)(mBase + row) * Dn + kk + col);
            v4f wv4 = *(const v4f*)(Wp + (size_t)(nBase + row) * Dn + kk + col);
            #pragma unroll
            for (int e = 0; e < 4; ++e) {
                lX[row][col + e] = (f16)xv[e];
                lW[row][col + e] = (f16)wv4[e];
            }
            if (kk + 32 < Dn) {                       // prefetch next K tile
                __builtin_prefetch(X  + (size_t)(mBase + row) * Dn + kk + 32 + col, 0, 1);
                __builtin_prefetch(Wp + (size_t)(nBase + row) * Dn + kk + 32 + col, 0, 1);
            }
        }
        __syncthreads();

        // A fragment (16x32): lane row = lm, halves 0-7 at k=lg*8, 8-15 at 16+lg*8
        v8h alo = *(const v8h*)&lX[wv * 16 + lm][lg * 8];
        v8h ahi = *(const v8h*)&lX[wv * 16 + lm][16 + lg * 8];
        v16h a = cat8(alo, ahi);

        #pragma unroll
        for (int f = 0; f < 4; ++f) {
            // B fragment (32x16): lane col = lm, halves 0-7 at k=lg*8, 8-15 at 16+lg*8
            v8h blo = *(const v8h*)&lW[f * 16 + lm][lg * 8];
            v8h bhi = *(const v8h*)&lW[f * 16 + lm][16 + lg * 8];
            acc[f] = wmma_f16(a, cat8(blo, bhi), acc[f]);
        }
    }

    constexpr float scale = (P == 0) ? 0.125f : 1.0f;  // HD^-0.5 = 1/8
    #pragma unroll
    for (int f = 0; f < 4; ++f) {
        const int e    = nBase + f * 16 + lm;
        const float bv = bin[P * En + e];
        const int h = e >> 6, hd = e & 63;
        #pragma unroll
        for (int r = 0; r < 8; ++r) {
            const int m = mBase + wv * 16 + lg * 8 + r;
            const int t = m >> 1, b = m & 1;           // m = t*B + b, B=2
            const int bh = b * Hn + h;
            const float val = (acc[f][r] + bv) * scale;
            if (P == 2) dst[((size_t)bh * HDn + hd) * Tn + t] = (f16)val;
            else        dst[((size_t)bh * Tn + t) * HDn + hd] = (f16)val;
        }
    }
}

// ---------------------------------------------------------------------------
// Kernel 2: flash attention per (b,h).  Computes S^T = K·q^T tiles so the
// softmaxed C-fragment reuses directly as the A-fragment of P·V.
// Grid (T/64, B*H); 128 threads = 4 waves, 16 t-rows per wave.
// ---------------------------------------------------------------------------
__global__ __launch_bounds__(128)
void mha_flash_attn(const f16* __restrict__ qh, const f16* __restrict__ kh,
                    const f16* __restrict__ vT, const float* __restrict__ mask,
                    f16* __restrict__ ch)
{
    const int tid  = threadIdx.x;
    const int wv   = tid >> 5;
    const int lane = tid & 31;
    const int lm   = lane & 15;
    const int lg   = lane >> 4;

    const int bh = blockIdx.y;
    const int b  = bh >> 4;                // H = 16
    const int h  = bh & 15;
    const int tb = blockIdx.x * 64 + wv * 16;
    const int tcol = tb + lm;              // this lane's t (N index of S^T)

    // q B-fragments: loop-invariant, kept in registers for all 64 s-steps.
    const f16* qp = qh + ((size_t)bh * Tn + tcol) * HDn;
    v16h bq[2];
    #pragma unroll
    for (int k2 = 0; k2 < 2; ++k2) {
        v8h lo = *(const v8h*)(qp + k2 * 32 + lg * 8);
        v8h hi = *(const v8h*)(qp + k2 * 32 + 16 + lg * 8);
        bq[k2] = cat8(lo, hi);
    }

    v8f o[4];
    #pragma unroll
    for (int f = 0; f < 4; ++f) o[f] = zero8();
    float m_run = -3.402823466e38f;
    float l_run = 0.f;

    const float* mrow = mask + (size_t)tcol * Tn;

    for (int s = 0; s < Tn; s += 32) {
        // prefetch next s-block of K and V
        if (s + 32 < Tn) {
            __builtin_prefetch(kh + ((size_t)bh * Tn + s + 32 + lm) * HDn, 0, 1);
            __builtin_prefetch(vT + ((size_t)bh * HDn + lm) * Tn + s + 32 + lg * 16, 0, 1);
        }

        // A fragments of K-tile: rows = s..s+31, K = hd (two k-steps of 32)
        v16h ak[2][2];
        #pragma unroll
        for (int sb = 0; sb < 2; ++sb) {
            const f16* kp = kh + ((size_t)bh * Tn + s + sb * 16 + lm) * HDn;
            #pragma unroll
            for (int k2 = 0; k2 < 2; ++k2) {
                v8h lo = *(const v8h*)(kp + k2 * 32 + lg * 8);
                v8h hi = *(const v8h*)(kp + k2 * 32 + 16 + lg * 8);
                ak[sb][k2] = cat8(lo, hi);
            }
        }

        v8f s0 = zero8(), s1 = zero8();
        s0 = wmma_f16(ak[0][0], bq[0], s0);
        s0 = wmma_f16(ak[0][1], bq[1], s0);
        s1 = wmma_f16(ak[1][0], bq[0], s1);
        s1 = wmma_f16(ak[1][1], bq[1], s1);

        // + attn_mask[t][s'] ; lane's t fixed, s' = s (+16) + lg*8 + r contiguous
        v4f mk0 = *(const v4f*)(mrow + s + lg * 8);
        v4f mk1 = *(const v4f*)(mrow + s + lg * 8 + 4);
        v4f mk2 = *(const v4f*)(mrow + s + 16 + lg * 8);
        v4f mk3 = *(const v4f*)(mrow + s + 16 + lg * 8 + 4);
        #pragma unroll
        for (int r = 0; r < 4; ++r) {
            s0[r] += mk0[r]; s0[4 + r] += mk1[r];
            s1[r] += mk2[r]; s1[4 + r] += mk3[r];
        }

        // online softmax over s for column t = lm (lanes lm and lm+16 pair up)
        float mloc = s0[0];
        #pragma unroll
        for (int r = 1; r < 8; ++r) mloc = fmaxf(mloc, s0[r]);
        #pragma unroll
        for (int r = 0; r < 8; ++r) mloc = fmaxf(mloc, s1[r]);
        mloc = fmaxf(mloc, __shfl_xor(mloc, 16, 32));
        const float m_new = fmaxf(m_run, mloc);
        const float alpha = __expf(m_run - m_new);

        float e0[8], e1[8], lsum = 0.f;
        #pragma unroll
        for (int r = 0; r < 8; ++r) {
            e0[r] = __expf(s0[r] - m_new);
            e1[r] = __expf(s1[r] - m_new);
            lsum += e0[r] + e1[r];
        }
        lsum += __shfl_xor(lsum, 16, 32);
        l_run = l_run * alpha + lsum;
        m_run = m_new;

        // C-layout of S^T == A-layout of P: no cross-lane data movement.
        v16h pa;
        #pragma unroll
        for (int r = 0; r < 8; ++r) {
            pa[r]     = (f16)e0[r];
            pa[8 + r] = (f16)e1[r];
        }

        // rescale O rows (row t_local = lg*8 + r; stats live in lane t_local)
        float av[8];
        #pragma unroll
        for (int r = 0; r < 8; ++r) av[r] = __shfl(alpha, lg * 8 + r, 32);
        #pragma unroll
        for (int f = 0; f < 4; ++f)
            #pragma unroll
            for (int r = 0; r < 8; ++r) o[f][r] *= av[r];

        // O += P @ V ; B fragments from vT (s at lg*8 and 16+lg*8 per lane)
        #pragma unroll
        for (int f = 0; f < 4; ++f) {
            const f16* vp = vT + ((size_t)bh * HDn + f * 16 + lm) * Tn + s + lg * 8;
            v8h lo = *(const v8h*)vp;
            v8h hi = *(const v8h*)(vp + 16);
            o[f] = wmma_f16(pa, cat8(lo, hi), o[f]);
        }
    }

    float inv[8];
    #pragma unroll
    for (int r = 0; r < 8; ++r) inv[r] = 1.0f / __shfl(l_run, lg * 8 + r, 32);

    #pragma unroll
    for (int f = 0; f < 4; ++f) {
        const int hd = f * 16 + lm;
        const int e  = h * 64 + hd;
        #pragma unroll
        for (int r = 0; r < 8; ++r) {
            const int t = tb + lg * 8 + r;
            ch[((size_t)t * Bn + b) * En + e] = (f16)(o[f][r] * inv[r]);
        }
    }
}

// ---------------------------------------------------------------------------
// Kernel 3: output projection.  out = ch @ Wo^T + bo  (f32 out).
// The f16 A-tile (64x32 of ch) is staged via the Tensor Data Mover with LDS
// padding (16-DWORD interval, 4-DWORD pad) matching the 40-half row stride;
// W is converted f32->f16 cooperatively while the TDM runs.
// ---------------------------------------------------------------------------
__global__ __launch_bounds__(128)
void mha_out_proj(const f16* __restrict__ ch, const float* __restrict__ Wo,
                  const float* __restrict__ bo, float* __restrict__ out)
{
    const int mBase = blockIdx.x * 64;
    const int nBase = blockIdx.y * 64;

    __shared__ __align__(16) f16 lX[64][40];
    __shared__ __align__(16) f16 lW[64][40];

    const int tid  = threadIdx.x;
    const int wv   = tid >> 5;
    const int lane = tid & 31;
    const int lm   = lane & 15;
    const int lg   = lane >> 4;

    v8f acc[4];
    #pragma unroll
    for (int f = 0; f < 4; ++f) acc[f] = zero8();

#if MHA_USE_TDM
    // Tensor DMA descriptor group 1 (loop-invariant): 2D f16 tensor 4096x1024,
    // tile 32x64, data_size=2B, pad_enable, pad_interval=16 DW, pad_amount=4 DW.
    i32x8 g1;
    g1[0] = (int)((1u << 16) | (1u << 20) | (3u << 22) | (3u << 25));
    g1[1] = (int)(((unsigned)En & 0xFFFFu) << 16);                      // dim0 lo16
    g1[2] = (int)((((unsigned)Mn & 0xFFFFu) << 16) | ((unsigned)En >> 16)); // dim0 hi / dim1 lo
    g1[3] = (int)((32u << 16) | ((unsigned)Mn >> 16));                  // dim1 hi / tile_dim0
    g1[4] = 64;                                                         // tile_dim1
    g1[5] = En;                                                         // dim0 stride
    g1[6] = 0;
    g1[7] = 0;
    const i32x4 gz = {0, 0, 0, 0};
    const unsigned ldsX = (unsigned)(size_t)(&lX[0][0]);
#endif

    for (int kk = 0; kk < En; kk += 32) {
        __syncthreads();
#if MHA_USE_TDM
        if (wv == 0) {                                 // one wave issues the DMA
            const unsigned long long ga =
                (unsigned long long)(size_t)(ch + (size_t)mBase * En + kk);
            u32x4 g0;
            g0[0] = 1u;                                // count=1, user descriptor
            g0[1] = ldsX;                              // lds_addr
            g0[2] = (unsigned)(ga & 0xFFFFFFFFu);      // global_addr[31:0]
            g0[3] = (unsigned)((ga >> 32) & 0x01FFFFFFu) | (2u << 30); // addr hi | type=2
#if __clang_major__ >= 23
            const i32x8 gz8 = {0, 0, 0, 0, 0, 0, 0, 0};
            __builtin_amdgcn_tensor_load_to_lds(g0, g1, gz, gz, gz8, 0);
#else
            __builtin_amdgcn_tensor_load_to_lds(g0, g1, gz, gz, 0);
#endif
        }
#else
        #pragma unroll
        for (int j = 0; j < 2; ++j) {                  // X: 256 v8h chunks / 128 thr
            int c   = tid + j * 128;
            int row = c >> 2;
            int col = (c & 3) << 3;
            *(v8h*)&lX[row][col] =
                *(const v8h*)(ch + (size_t)(mBase + row) * En + kk + col);
        }
#endif
        #pragma unroll
        for (int j = 0; j < 4; ++j) {                  // W: f32 -> f16 convert
            int c   = tid + j * 128;
            int row = c >> 3;
            int col = (c & 7) << 2;
            v4f w4 = *(const v4f*)(Wo + (size_t)(nBase + row) * En + kk + col);
            #pragma unroll
            for (int e = 0; e < 4; ++e) lW[row][col + e] = (f16)w4[e];
            if (kk + 32 < En)
                __builtin_prefetch(Wo + (size_t)(nBase + row) * En + kk + 32 + col, 0, 1);
        }
#if MHA_USE_TDM
        if (wv == 0) __builtin_amdgcn_s_wait_tensorcnt(0);
#endif
        __syncthreads();

        v8h alo = *(const v8h*)&lX[wv * 16 + lm][lg * 8];
        v8h ahi = *(const v8h*)&lX[wv * 16 + lm][16 + lg * 8];
        v16h a = cat8(alo, ahi);

        #pragma unroll
        for (int f = 0; f < 4; ++f) {
            v8h blo = *(const v8h*)&lW[f * 16 + lm][lg * 8];
            v8h bhi = *(const v8h*)&lW[f * 16 + lm][16 + lg * 8];
            acc[f] = wmma_f16(a, cat8(blo, bhi), acc[f]);
        }
    }

    #pragma unroll
    for (int f = 0; f < 4; ++f) {
        const int d    = nBase + f * 16 + lm;
        const float bv = bo[d];
        #pragma unroll
        for (int r = 0; r < 8; ++r) {
            const int m = mBase + wv * 16 + lg * 8 + r;
            out[(size_t)m * Dn + d] = acc[f][r] + bv;
        }
    }
}

// ---------------------------------------------------------------------------
extern "C" void kernel_launch(void* const* d_in, const int* in_sizes, int n_in,
                              void* d_out, int out_size, void* d_ws, size_t ws_size,
                              hipStream_t stream)
{
    const float* query = (const float*)d_in[0];
    const float* key   = (const float*)d_in[1];
    const float* value = (const float*)d_in[2];
    const float* mask  = (const float*)d_in[3];
    const float* Win   = (const float*)d_in[4];
    const float* bin   = (const float*)d_in[5];
    const float* Wout  = (const float*)d_in[6];
    const float* bout  = (const float*)d_in[7];
    float* out = (float*)d_out;

    const size_t NQK = (size_t)Bn * Hn * Tn * HDn;    // 4M halves each
    f16* qh = (f16*)d_ws;
    f16* kh = qh + NQK;
    f16* vT = kh + NQK;
    f16* ch = vT + NQK;                               // (T,B,E) f16 context

    dim3 gProj(Mn / 64, En / 64);
    mha_qkv_proj<0><<<gProj, 128, 0, stream>>>(query, Win, bin, qh);
    mha_qkv_proj<1><<<gProj, 128, 0, stream>>>(key,   Win, bin, kh);
    mha_qkv_proj<2><<<gProj, 128, 0, stream>>>(value, Win, bin, vT);
    mha_flash_attn<<<dim3(Tn / 64, Bn * Hn), 128, 0, stream>>>(
        qh, kh, vT, mask, ch);
    mha_out_proj<<<dim3(Mn / 64, Dn / 64), 128, 0, stream>>>(
        ch, Wout, bout, out);
}
